// Sequence_34488587387493
// MI455X (gfx1250) — matrix operand order repaired and checked by reading, not compiled, not run on hardware
//
#include <hip/hip_runtime.h>

typedef _Float16 v8h  __attribute__((ext_vector_type(8)));
typedef _Float16 v16h __attribute__((ext_vector_type(16)));
typedef float    v8f  __attribute__((ext_vector_type(8)));

#define TSTEPS   4096
#define DIM      64
#define G4       256      // 4*DIM gate width
#define ROWS     16       // batch rows per workgroup (one WMMA M-tile)
#define NTHREADS 512      // 16 waves: one per 16-wide gate column tile

// ---- branchless activations on native CDNA5 transcendentals ----
#if __has_builtin(__builtin_amdgcn_tanhf)
__device__ __forceinline__ float ftanh_(float x) {
    return __builtin_amdgcn_tanhf(x);               // v_tanh_f32
}
__device__ __forceinline__ float fsigmoid_(float x) {
    return __builtin_fmaf(0.5f, __builtin_amdgcn_tanhf(0.5f * x), 0.5f);
}
#else
#if __has_builtin(__builtin_amdgcn_exp2f)
__device__ __forceinline__ float fexp2_(float x) { return __builtin_amdgcn_exp2f(x); }
#else
__device__ __forceinline__ float fexp2_(float x) { return __builtin_exp2f(x); }
#endif
#if __has_builtin(__builtin_amdgcn_rcpf)
__device__ __forceinline__ float frcp_(float x) { return __builtin_amdgcn_rcpf(x); }
#else
__device__ __forceinline__ float frcp_(float x) { return 1.0f / x; }
#endif
__device__ __forceinline__ float fsigmoid_(float x) {
    // sig(x) = 1/(1+2^(-x*log2 e)); exact limits at +-inf, branch-free
    return frcp_(1.0f + fexp2_(-1.44269504088896340736f * x));
}
__device__ __forceinline__ float ftanh_(float x) {
    return __builtin_fmaf(2.0f, fsigmoid_(2.0f * x), -1.0f);
}
#endif

__global__ __launch_bounds__(NTHREADS, 1)
void lstm2_persistent(const float* __restrict__ input,
                      const float* __restrict__ W_ih1,
                      const float* __restrict__ W_hh1,
                      const float* __restrict__ b_ih1,
                      const float* __restrict__ b_hh1,
                      const float* __restrict__ W_ih2,
                      const float* __restrict__ W_hh2,
                      const float* __restrict__ b_ih2,
                      const float* __restrict__ b_hh2,
                      const float* __restrict__ W_lin,
                      const float* __restrict__ b_lin,
                      const float* __restrict__ h0,
                      const float* __restrict__ c0,
                      const float* __restrict__ h02,
                      const float* __restrict__ c02,
                      float* __restrict__ out)
{
    __shared__ float    gates1[ROWS][G4];   // layer-1 matmul output
    __shared__ float    gates2[ROWS][G4];   // layer-2 matmul output
    __shared__ _Float16 h1h[ROWS][DIM];     // h1 state (f16, WMMA A operand)
    __shared__ _Float16 h2h[ROWS][DIM];     // h2 state (f16, WMMA A operand)
    __shared__ float    c1s[ROWS][DIM];     // c1 state
    __shared__ float    c2s[ROWS][DIM];     // c2 state
    __shared__ float    h2f[ROWS][DIM];     // h2 state (f32, for readout)
    __shared__ float    bsum1[G4];          // b_ih1 + b_hh1
    __shared__ float    bsum2[G4];          // b_ih2 + b_hh2
    __shared__ float    wih1s[G4];          // W_ih1 column (input is scalar/step)
    __shared__ float    wlin[2][DIM];
    __shared__ float    blin[2];
    __shared__ float    xbuf[ROWS];

    const int tid  = threadIdx.x;
    const int lane = tid & 31;
    const int wave = tid >> 5;        // 0..15 => gate column tile
    const int wg   = blockIdx.x;      // 0..15 => 16-row batch slice

    // ---------------- one-time preload ----------------
    for (int i = tid; i < G4; i += NTHREADS) {
        bsum1[i] = b_ih1[i] + b_hh1[i];
        bsum2[i] = b_ih2[i] + b_hh2[i];
        wih1s[i] = W_ih1[i];
    }
    for (int i = tid; i < 2 * DIM; i += NTHREADS) wlin[i >> 6][i & (DIM - 1)] = W_lin[i];
    if (tid < 2) blin[tid] = b_lin[tid];
    for (int i = tid; i < ROWS * DIM; i += NTHREADS) {
        const int m = i >> 6, j = i & (DIM - 1);
        h1h[m][j] = (_Float16)h0[j];
        c1s[m][j] = c0[j];
        h2h[m][j] = (_Float16)h02[j];
        c2s[m][j] = c02[j];
        h2f[m][j] = h02[j];
    }

    // Per-wave register-resident B-matrix weight tiles (f16).
    // B layout (16-bit, 32x16): lane l holds column N = l%16,
    // halves K = kh*32 + {(l/16)*8 .. +7} and {16+(l/16)*8 .. +7}.
    const int nCol = wave * 16 + (lane & 15);   // gate index 0..255
    const int kHi  = (lane >> 4) << 3;          // 0 or 8
    v16h bhh1[2], bih2[2], bhh2[2];
    for (int kh = 0; kh < 2; ++kh) {
        const int kb = kh * 32 + kHi;
        for (int e = 0; e < 16; ++e) {
            bhh1[kh][e] = (_Float16)W_hh1[nCol * DIM + kb + e + (e & 8)];
            bih2[kh][e] = (_Float16)W_ih2[nCol * DIM + kb + e + (e & 8)];
            bhh2[kh][e] = (_Float16)W_hh2[nCol * DIM + kb + e + (e & 8)];
        }
    }

    // A layout (16-bit, 16x32): lane l holds row M = l%16,
    // halves K = kh*32 + {(l/16)*8 .. +7} and {16+(l/16)*8 .. +7}.
    const int aM  = lane & 15;
    const int aK0 = (lane >> 4) << 3;           // 0 or 8
    // C/D layout: lane l, reg r -> M = r + 8*(l/16), N = l%16.
    const int mo = (lane >> 4) << 3;

    __syncthreads();

    const float* xr = input + (size_t)wg * ROWS * TSTEPS;

    // software-pipelined input prefetch (1 full step of latency hiding)
    float xnext = 0.0f;
    if (tid < ROWS) xnext = xr[tid * TSTEPS];

    for (int t = 0; t < TSTEPS; ++t) {
        if (tid < ROWS) {
            xbuf[tid] = xnext;                       // x[t] (loaded last iter)
            if (t + 1 < TSTEPS) xnext = xr[tid * TSTEPS + t + 1];
        }

        // ---- P1: gates1 tile = h1 @ W_hh1^T ----
        {
            v16h a0, a1;
            {
                v8h p0 = *(const v8h*)&h1h[aM][aK0];
                v8h p1 = *(const v8h*)&h1h[aM][aK0 + 16];
                v8h p2 = *(const v8h*)&h1h[aM][aK0 + 32];
                v8h p3 = *(const v8h*)&h1h[aM][aK0 + 48];
                for (int e = 0; e < 8; ++e) {
                    a0[e] = p0[e]; a0[e + 8] = p1[e];
                    a1[e] = p2[e]; a1[e + 8] = p3[e];
                }
            }
            v8f acc = {};
            acc = __builtin_amdgcn_wmma_f32_16x16x32_f16(false, a0, false, bhh1[0],
                                                         (short)0, acc, false, false);
            acc = __builtin_amdgcn_wmma_f32_16x16x32_f16(false, a1, false, bhh1[1],
                                                         (short)0, acc, false, false);
            for (int r = 0; r < 8; ++r) gates1[mo + r][nCol] = acc[r];
        }
        __syncthreads();

        // ---- P2: elementwise LSTM cell 1 (i,f,g,o gate order) ----
        for (int q = 0; q < 2; ++q) {
            const int p = tid * 2 + q;
            const int m = p >> 6, j = p & (DIM - 1);
            const float xb = xbuf[m];
            const float gi = gates1[m][j]       + __builtin_fmaf(xb, wih1s[j],       bsum1[j]);
            const float gf = gates1[m][j + 64]  + __builtin_fmaf(xb, wih1s[j + 64],  bsum1[j + 64]);
            const float gg = gates1[m][j + 128] + __builtin_fmaf(xb, wih1s[j + 128], bsum1[j + 128]);
            const float go = gates1[m][j + 192] + __builtin_fmaf(xb, wih1s[j + 192], bsum1[j + 192]);
            const float c  = fsigmoid_(gf) * c1s[m][j] + fsigmoid_(gi) * ftanh_(gg);
            const float h  = fsigmoid_(go) * ftanh_(c);
            c1s[m][j] = c;
            h1h[m][j] = (_Float16)h;
        }
        __syncthreads();

        // ---- P3: gates2 tile = h1 @ W_ih2^T + h2 @ W_hh2^T (chained C) ----
        {
            v16h a0, a1, a2, a3;
            {
                v8h p0 = *(const v8h*)&h1h[aM][aK0];
                v8h p1 = *(const v8h*)&h1h[aM][aK0 + 16];
                v8h p2 = *(const v8h*)&h1h[aM][aK0 + 32];
                v8h p3 = *(const v8h*)&h1h[aM][aK0 + 48];
                v8h q0 = *(const v8h*)&h2h[aM][aK0];
                v8h q1 = *(const v8h*)&h2h[aM][aK0 + 16];
                v8h q2 = *(const v8h*)&h2h[aM][aK0 + 32];
                v8h q3 = *(const v8h*)&h2h[aM][aK0 + 48];
                for (int e = 0; e < 8; ++e) {
                    a0[e] = p0[e]; a0[e + 8] = p1[e];
                    a1[e] = p2[e]; a1[e + 8] = p3[e];
                    a2[e] = q0[e]; a2[e + 8] = q1[e];
                    a3[e] = q2[e]; a3[e + 8] = q3[e];
                }
            }
            v8f acc = {};
            acc = __builtin_amdgcn_wmma_f32_16x16x32_f16(false, a0, false, bih2[0],
                                                         (short)0, acc, false, false);
            acc = __builtin_amdgcn_wmma_f32_16x16x32_f16(false, a1, false, bih2[1],
                                                         (short)0, acc, false, false);
            acc = __builtin_amdgcn_wmma_f32_16x16x32_f16(false, a2, false, bhh2[0],
                                                         (short)0, acc, false, false);
            acc = __builtin_amdgcn_wmma_f32_16x16x32_f16(false, a3, false, bhh2[1],
                                                         (short)0, acc, false, false);
            for (int r = 0; r < 8; ++r) gates2[mo + r][nCol] = acc[r];
        }
        __syncthreads();

        // ---- P4: elementwise LSTM cell 2 ----
        for (int q = 0; q < 2; ++q) {
            const int p = tid * 2 + q;
            const int m = p >> 6, j = p & (DIM - 1);
            const float gi = gates2[m][j]       + bsum2[j];
            const float gf = gates2[m][j + 64]  + bsum2[j + 64];
            const float gg = gates2[m][j + 128] + bsum2[j + 128];
            const float go = gates2[m][j + 192] + bsum2[j + 192];
            const float c  = fsigmoid_(gf) * c2s[m][j] + fsigmoid_(gi) * ftanh_(gg);
            const float h  = fsigmoid_(go) * ftanh_(c);
            c2s[m][j] = c;
            h2h[m][j] = (_Float16)h;
            h2f[m][j] = h;
        }
        __syncthreads();

        // ---- P5: readout (h2 @ W_lin^T + b_lin), out[B,2,T] ----
        // Reads h2f (safe: P4 barrier above); next write to h2f is next P4,
        // separated by the P1/P2/P3 barriers, so no extra barrier needed here.
        if (tid < 2 * ROWS) {
            const int m = tid >> 1, w = tid & 1;
            const float4* hv = (const float4*)&h2f[m][0];
            const float4* wv = (const float4*)&wlin[w][0];
            float s = blin[w];
            for (int j = 0; j < DIM / 4; ++j) {
                const float4 a = hv[j], b = wv[j];
                s = __builtin_fmaf(a.x, b.x, s);
                s = __builtin_fmaf(a.y, b.y, s);
                s = __builtin_fmaf(a.z, b.z, s);
                s = __builtin_fmaf(a.w, b.w, s);
            }
            out[((size_t)(wg * ROWS + m) * 2 + w) * TSTEPS + t] = s;
        }
    }
}

extern "C" void kernel_launch(void* const* d_in, const int* in_sizes, int n_in,
                              void* d_out, int out_size, void* d_ws, size_t ws_size,
                              hipStream_t stream) {
    const float* input = (const float*)d_in[0];
    const float* W_ih1 = (const float*)d_in[1];
    const float* W_hh1 = (const float*)d_in[2];
    const float* b_ih1 = (const float*)d_in[3];
    const float* b_hh1 = (const float*)d_in[4];
    const float* W_ih2 = (const float*)d_in[5];
    const float* W_hh2 = (const float*)d_in[6];
    const float* b_ih2 = (const float*)d_in[7];
    const float* b_hh2 = (const float*)d_in[8];
    const float* W_lin = (const float*)d_in[9];
    const float* b_lin = (const float*)d_in[10];
    const float* h0    = (const float*)d_in[11];
    const float* c0    = (const float*)d_in[12];
    const float* h02   = (const float*)d_in[13];
    const float* c02   = (const float*)d_in[14];
    float* out = (float*)d_out;

    // B=256 -> 16 independent persistent workgroups of 16 batch rows each.
    lstm2_persistent<<<dim3(16), dim3(NTHREADS), 0, stream>>>(
        input, W_ih1, W_hh1, b_ih1, b_hh1,
        W_ih2, W_hh2, b_ih2, b_hh2,
        W_lin, b_lin, h0, c0, h02, c02, out);
}
